// HungarianVQVAE_46248207843397
// MI455X (gfx1250) — compile-verified, matrix operand order checked by reading, not run on hardware
//
#include <hip/hip_runtime.h>
#include <hip/hip_bf16.h>
#include <math.h>

// ---------------------------------------------------------------------------
// Problem dims (fixed by the reference)
// ---------------------------------------------------------------------------
#define BATCH        512
#define HIDDEN_DIM   4096
#define NUM_CODES    32
#define CODE_DIM     64
#define CODEBOOK_SZ  1024
#define ENC_OUT      (NUM_CODES * CODE_DIM)     // 2048
#define EF_ROWS      (BATCH * NUM_CODES)        // 16384

// ---------------------------------------------------------------------------
// CDNA5 WMMA / TDM types
// ---------------------------------------------------------------------------
typedef __attribute__((ext_vector_type(16))) __bf16 v16bf;
typedef __attribute__((ext_vector_type(8)))  float  v8f;
typedef __attribute__((ext_vector_type(4))) unsigned int uint32x4;
typedef __attribute__((ext_vector_type(8))) int          int32x8;
typedef __attribute__((ext_vector_type(4))) int          int32x4;

union FragBF {
    v16bf v;
    uint4 q[2];
};

__device__ __forceinline__ unsigned short f32_to_bf16_bits(float f) {
    unsigned int u = __float_as_uint(f);
    unsigned int r = u + 0x7FFFu + ((u >> 16) & 1u);  // round-to-nearest-even
    return (unsigned short)(r >> 16);
}

#define WMMA_BF16(A, B, C)                                                     \
    __builtin_amdgcn_wmma_f32_16x16x32_bf16(false, (A), false, (B), (short)0,  \
                                            (C), false, false)

// Low 32 bits of a flat pointer into the LDS aperture = LDS byte offset
__device__ __forceinline__ unsigned int lds_off(const void* p) {
    return (unsigned int)(size_t)p;
}

// ---------------------------------------------------------------------------
// TDM: issue TENSOR_LOAD_TO_LDS of a 2D bf16 tile [tile_rows x tile_k] from a
// row-major [*, stride_elems] bf16 matrix starting at gptr, into a
// contiguous LDS tile at lds_byte (row-major [tile_rows][tile_k]).
// D# layout per CDNA5 ISA (6-arg builtin form).
// ---------------------------------------------------------------------------
__device__ __forceinline__ void tdm_load_2d_bf16(
    unsigned int lds_byte, const unsigned short* gptr,
    unsigned int tensor_d0, unsigned int tensor_d1,
    unsigned int tile_k, unsigned int tile_rows,
    unsigned int stride_elems) {
    unsigned long long ga = (unsigned long long)(size_t)gptr;
    uint32x4 g0;
    g0[0] = 1u;                                   // count=1 (valid), gather off
    g0[1] = lds_byte;                             // lds_addr  (bits 63:32)
    g0[2] = (unsigned int)(ga & 0xFFFFFFFFu);     // global_addr[31:0]
    g0[3] = (unsigned int)((ga >> 32) & 0x01FFFFFFu)  // global_addr[56:32]
          | 0x80000000u;                          // type=2 ("image", bits 127:126)
    int32x8 g1;
    g1[0] = (int)(1u << 16);                      // wg_mask=0, data_size=1 (2B)
    g1[1] = (int)((tensor_d0 & 0xFFFFu) << 16);   // tensor_dim0[15:0] @ bits 63:48
    g1[2] = (int)((tensor_d0 >> 16) & 0xFFFFu)    // tensor_dim0[31:16]
          | (int)((tensor_d1 & 0xFFFFu) << 16);   // tensor_dim1[15:0]
    g1[3] = (int)((tensor_d1 >> 16) & 0xFFFFu)    // tensor_dim1[31:16]
          | (int)((tile_k & 0xFFFFu) << 16);      // tile_dim0 @ bits 127:112
    g1[4] = (int)(tile_rows & 0xFFFFu);           // tile_dim1; tile_dim2=0
    g1[5] = (int)stride_elems;                    // tensor_dim0_stride[31:0]
    g1[6] = 0;                                    // stride[47:32]=0, dim1_stride lo=0
    g1[7] = 0;                                    // dim1_stride hi=0
    int32x4 z4 = {0, 0, 0, 0};
    int32x8 z8 = {0, 0, 0, 0, 0, 0, 0, 0};
    __builtin_amdgcn_tensor_load_to_lds(g0, g1, z4, z4, z8, 0);
}

// Fragment (16x32 bf16 A/B operand) from a contiguous LDS tile [rows][32]:
// lane L holds row (L&15)(+half), K chunks at (L>>4)*8 and +16 (2x b128 ds).
__device__ __forceinline__ v16bf lds_frag(const unsigned short* tile, int row,
                                          int lane) {
    int koff = (lane >> 4) * 8;
    const unsigned short* rp = tile + row * 32 + koff;
    FragBF f;
    f.q[0] = *(const uint4*)(rp);
    f.q[1] = *(const uint4*)(rp + 16);
    return f.v;
}

// Fragment straight from global (row-major [rows][ldk] bf16)
__device__ __forceinline__ v16bf load_frag(const unsigned short* base,
                                           int row, int ldk, int k0, int lane) {
    int koff = (lane >> 4) * 8;
    const unsigned short* rp = base + (size_t)row * (size_t)ldk + k0 + koff;
    FragBF f;
    f.q[0] = *(const uint4*)(rp);
    f.q[1] = *(const uint4*)(rp + 16);
    return f.v;
}

// One K=32 step of the 32x64 tile: 8 WMMAs, interleaved so at most
// a0,a1 + one B fragment are live (keeps regalloc out of scratch).
__device__ __forceinline__ void gemm_step(const unsigned short* tA,
                                          const unsigned short* tB,
                                          v8f* acc0, v8f* acc1,
                                          int rA, int lane) {
    v16bf a0 = lds_frag(tA, rA, lane);
    v16bf a1 = lds_frag(tA, rA + 16, lane);
#pragma unroll
    for (int j = 0; j < 4; ++j) {
        v16bf b = lds_frag(tB, rA + j * 16, lane);
        acc0[j] = WMMA_BF16(a0, b, acc0[j]);
        acc1[j] = WMMA_BF16(a1, b, acc1[j]);
    }
}

// ---------------------------------------------------------------------------
// fp32 -> bf16 convert, row norms
// ---------------------------------------------------------------------------
__global__ void k_f32_to_bf16(const float* __restrict__ in,
                              unsigned short* __restrict__ out, int n) {
    int i = blockIdx.x * blockDim.x + threadIdx.x;
    if (i < n) out[i] = f32_to_bf16_bits(in[i]);
}

__global__ void k_rownorm(const float* __restrict__ in,
                          float* __restrict__ norms, int rows, int cols) {
    int r = blockIdx.x * blockDim.x + threadIdx.x;
    if (r >= rows) return;
    const float* p = in + (size_t)r * (size_t)cols;
    float s = 0.f;
    for (int c = 0; c < cols; ++c) { float t = p[c]; s = fmaf(t, t, s); }
    norms[r] = s;
}

// ---------------------------------------------------------------------------
// WMMA GEMM with TDM double-buffered LDS staging:
//   C[M,N] = A[M,K](bf16) * B[N,K](bf16)^T + bias[N]
// One wave32 per 32x64 C tile (2x4 fragments = 8 wmma per K=32 step).
// Steady-state loop: issue 2 TDM loads -> s_wait_tensorcnt(2) -> 8 WMMAs.
// __launch_bounds__(32,1): single wave owns the register file (no spills).
// ---------------------------------------------------------------------------
__global__ __launch_bounds__(32, 1)
void k_gemm_bf16_nt_tdm(const unsigned short* __restrict__ A,
                        const unsigned short* __restrict__ B,
                        const float* __restrict__ bias,
                        float* __restrict__ C,
                        int M, int N, int K) {
    __shared__ unsigned short sA[2][32 * 32];   // 2 x 2 KB
    __shared__ unsigned short sB[2][64 * 32];   // 2 x 4 KB

    int lane  = threadIdx.x & 31;
    int nBase = blockIdx.x * 64;
    int mBase = blockIdx.y * 32;
    int rA    = lane & 15;

    const unsigned short* Abase = A + (size_t)mBase * K;
    const unsigned short* Bbase = B + (size_t)nBase * K;

    // stage 0 DMA
    tdm_load_2d_bf16(lds_off(&sA[0][0]), Abase,
                     (unsigned)K, (unsigned)(M - mBase), 32, 32, (unsigned)K);
    tdm_load_2d_bf16(lds_off(&sB[0][0]), Bbase,
                     (unsigned)K, (unsigned)(N - nBase), 32, 64, (unsigned)K);

    v8f acc0[4], acc1[4];
#pragma unroll
    for (int j = 0; j < 4; ++j) { acc0[j] = (v8f){}; acc1[j] = (v8f){}; }

    int nk = K / 32;
    for (int s = 0; s < nk - 1; ++s) {
        int cur = s & 1;
        int nxt = cur ^ 1;
        int k0  = (s + 1) * 32;
        tdm_load_2d_bf16(lds_off(&sA[nxt][0]), Abase + k0,
                         (unsigned)(K - k0), (unsigned)(M - mBase), 32, 32,
                         (unsigned)K);
        tdm_load_2d_bf16(lds_off(&sB[nxt][0]), Bbase + k0,
                         (unsigned)(K - k0), (unsigned)(N - nBase), 32, 64,
                         (unsigned)K);
        // TDM completes in-order: <=2 outstanding => current stage landed
        __builtin_amdgcn_s_wait_tensorcnt(2);
        gemm_step(&sA[cur][0], &sB[cur][0], acc0, acc1, rA, lane);
    }
    __builtin_amdgcn_s_wait_tensorcnt(0);
    gemm_step(&sA[(nk - 1) & 1][0], &sB[(nk - 1) & 1][0], acc0, acc1, rA, lane);

    // C/D layout: lane L holds N = L&15, M = v + 8*(L>>4) for VGPR v in 0..7
    int nC   = lane & 15;
    int mOff = (lane >> 4) * 8;
#pragma unroll
    for (int j = 0; j < 4; ++j) {
        int n  = nBase + j * 16 + nC;
        float bn = bias[n];
#pragma unroll
        for (int v = 0; v < 8; ++v) {
            int m0 = mBase + mOff + v;
            C[(size_t)m0 * N + n]        = acc0[j][v] + bn;
            C[(size_t)(m0 + 16) * N + n] = acc1[j][v] + bn;
        }
    }
}

// ---------------------------------------------------------------------------
// Distance kernel: dist[m][n] = sqrt(max(enorm[m] + cnorm[n] - 2*(ef.cb^T), 0))
// M = 16384 ef rows, N = 1024 codebook rows, K = 64 (L2-resident operands).
// ---------------------------------------------------------------------------
__global__ __launch_bounds__(32, 1)
void k_dist(const unsigned short* __restrict__ eb,
            const unsigned short* __restrict__ cbb,
            const float* __restrict__ enorm,
            const float* __restrict__ cnorm,
            float* __restrict__ dist) {
    const int K = CODE_DIM, N = CODEBOOK_SZ;
    int lane  = threadIdx.x & 31;
    int nBase = blockIdx.x * 32;
    int mBase = blockIdx.y * 32;
    int rA    = lane & 15;

    v8f acc00 = {}, acc01 = {}, acc10 = {}, acc11 = {};
#pragma unroll
    for (int k0 = 0; k0 < K; k0 += 32) {
        v16bf a0 = load_frag(eb,  mBase + rA,      K, k0, lane);
        v16bf a1 = load_frag(eb,  mBase + 16 + rA, K, k0, lane);
        v16bf b0 = load_frag(cbb, nBase + rA,      K, k0, lane);
        v16bf b1 = load_frag(cbb, nBase + 16 + rA, K, k0, lane);
        acc00 = WMMA_BF16(a0, b0, acc00);
        acc01 = WMMA_BF16(a0, b1, acc01);
        acc10 = WMMA_BF16(a1, b0, acc10);
        acc11 = WMMA_BF16(a1, b1, acc11);
    }

    int nC   = lane & 15;
    int mOff = (lane >> 4) * 8;
    float cn0 = cnorm[nBase + nC];
    float cn1 = cnorm[nBase + 16 + nC];
#pragma unroll
    for (int v = 0; v < 8; ++v) {
        int m0 = mBase + mOff + v;
        int m1 = m0 + 16;
        float em0 = enorm[m0], em1 = enorm[m1];
        float d00 = em0 + cn0 - 2.0f * acc00[v];
        float d01 = em0 + cn1 - 2.0f * acc01[v];
        float d10 = em1 + cn0 - 2.0f * acc10[v];
        float d11 = em1 + cn1 - 2.0f * acc11[v];
        dist[(size_t)m0 * N + nBase + nC]      = sqrtf(fmaxf(d00, 0.f));
        dist[(size_t)m0 * N + nBase + 16 + nC] = sqrtf(fmaxf(d01, 0.f));
        dist[(size_t)m1 * N + nBase + nC]      = sqrtf(fmaxf(d10, 0.f));
        dist[(size_t)m1 * N + nBase + 16 + nC] = sqrtf(fmaxf(d11, 0.f));
    }
}

// ---------------------------------------------------------------------------
// Hungarian (Jonker-Volgenant shortest augmenting path), one wave32 / sample.
// cost: 32 x 1024. Lane L owns columns j = 1+L, 1+L+32, ... (32 per lane).
// ---------------------------------------------------------------------------
#define HM NUM_CODES
#define HN CODEBOOK_SZ

__global__ __launch_bounds__(32)
void k_hungarian(const float* __restrict__ dist,
                 int* __restrict__ idx_out) {
    int s    = blockIdx.x;
    int lane = threadIdx.x;
    const float* cost = dist + (size_t)s * HM * HN;

    __shared__ float         v_[HN + 1];
    __shared__ float         minv[HN + 1];
    __shared__ int           way[HN + 1];
    __shared__ int           p[HN + 1];
    __shared__ unsigned char used[HN + 1];
    __shared__ float         u_[HM + 1];
    __shared__ int           s_i0;

    for (int j = lane; j <= HN; j += 32) { v_[j] = 0.f; p[j] = 0; }
    u_[lane] = 0.f;
    if (lane == 0) u_[HM] = 0.f;
    __syncthreads();

    const float INF = __builtin_inff();

    for (int i = 1; i <= HM; ++i) {
        for (int j = lane; j <= HN; j += 32) {
            minv[j] = INF; way[j] = 0; used[j] = 0;
        }
        if (lane == 0) p[0] = i;
        __syncthreads();

        int j0 = 0;
        while (true) {
            if (lane == 0) { used[j0] = 1; s_i0 = p[j0]; }
            __syncthreads();
            int   i0  = s_i0;
            float ui0 = u_[i0];
            const float* crow = cost + (size_t)(i0 - 1) * HN;

            float bmin = INF;
            int   bj   = 0x7FFFFFFF;
#pragma unroll 4
            for (int t = 0; t < HN / 32; ++t) {
                int j = 1 + lane + t * 32;
                if (!used[j]) {
                    float cur = crow[j - 1] - ui0 - v_[j];
                    if (cur < minv[j]) { minv[j] = cur; way[j] = j0; }
                    float mj = minv[j];
                    if (mj < bmin) { bmin = mj; bj = j; }
                }
            }
            // wave32 min+argmin reduction, tie-break on smaller column index
#pragma unroll
            for (int off = 16; off > 0; off >>= 1) {
                float om = __shfl_xor(bmin, off, 32);
                int   oj = __shfl_xor(bj, off, 32);
                if (om < bmin || (om == bmin && oj < bj)) { bmin = om; bj = oj; }
            }
            float delta = bmin;
            int   k     = bj;
            __syncthreads();

            // dual updates: distinct p[j] per used column -> race-free
#pragma unroll 4
            for (int t = 0; t < HN / 32; ++t) {
                int j = 1 + lane + t * 32;
                if (used[j]) {
                    u_[p[j]] += delta;
                    v_[j]    -= delta;
                } else {
                    minv[j]  -= delta;
                }
            }
            if (lane == 0) u_[p[0]] += delta;  // column 0 is always "used"
            __syncthreads();

            j0 = k;
            if (p[j0] == 0) break;
        }

        // augment along alternating path (serial, uniform j0 on all lanes)
        if (lane == 0) {
            int jj = j0;
            while (jj != 0) {
                int j1 = way[jj];
                p[jj]  = p[j1];
                jj     = j1;
            }
        }
        __syncthreads();
    }

    for (int j = 1 + lane; j <= HN; j += 32) {
        int pi = p[j];
        if (pi > 0) idx_out[(size_t)s * HM + (pi - 1)] = j - 1;
    }
}

// ---------------------------------------------------------------------------
// Gather: q = codebook[flat_idx] (fp32 out + bf16 scratch), export flat_idx
// ---------------------------------------------------------------------------
__global__ void k_gather(const float* __restrict__ codebook,
                         const int* __restrict__ idx,
                         float* __restrict__ q_out,
                         unsigned short* __restrict__ qb,
                         int* __restrict__ idx_out, int nrows) {
    int t = blockIdx.x * blockDim.x + threadIdx.x;
    if (t >= nrows * CODE_DIM) return;
    int row = t >> 6;
    int c   = t & (CODE_DIM - 1);
    int id  = idx[row];
    float val = codebook[(size_t)id * CODE_DIM + c];
    q_out[t] = val;
    qb[t]    = f32_to_bf16_bits(val);
    if (c == 0) idx_out[row] = id;
}

// ---------------------------------------------------------------------------
// Launcher
// ---------------------------------------------------------------------------
extern "C" void kernel_launch(void* const* d_in, const int* in_sizes, int n_in,
                              void* d_out, int out_size, void* d_ws, size_t ws_size,
                              hipStream_t stream) {
    (void)in_sizes; (void)n_in; (void)out_size; (void)ws_size;

    const float* x        = (const float*)d_in[0];  // [512,4096]
    const float* codebook = (const float*)d_in[1];  // [1024,64]
    const float* enc_w    = (const float*)d_in[2];  // [2048,4096]
    const float* enc_b    = (const float*)d_in[3];  // [2048]
    const float* dec_w    = (const float*)d_in[4];  // [4096,2048]
    const float* dec_b    = (const float*)d_in[5];  // [4096]

    // output slots: r | q | e | flat_idx
    float* out    = (float*)d_out;
    float* r_out  = out;                                   // 512*4096
    float* q_out  = r_out + (size_t)BATCH * HIDDEN_DIM;    // 512*2048
    float* e_out  = q_out + (size_t)BATCH * ENC_OUT;       // 512*2048
    int*   fi_out = (int*)(e_out + (size_t)BATCH * ENC_OUT);

    // workspace layout (16B-aligned slabs)
    char* w = (char*)d_ws;
    unsigned short* xb   = (unsigned short*)w; w += (size_t)BATCH * HIDDEN_DIM * 2;      // 4 MB
    unsigned short* ewb  = (unsigned short*)w; w += (size_t)ENC_OUT * HIDDEN_DIM * 2;    // 16 MB
    unsigned short* dwb  = (unsigned short*)w; w += (size_t)HIDDEN_DIM * ENC_OUT * 2;    // 16 MB
    unsigned short* cbb  = (unsigned short*)w; w += (size_t)CODEBOOK_SZ * CODE_DIM * 2;  // 128 KB
    unsigned short* ebuf = (unsigned short*)w; w += (size_t)EF_ROWS * CODE_DIM * 2;      // 2 MB
    unsigned short* qb   = (unsigned short*)w; w += (size_t)BATCH * ENC_OUT * 2;         // 2 MB
    float* cnorm = (float*)w; w += (size_t)CODEBOOK_SZ * 4;
    float* enorm = (float*)w; w += (size_t)EF_ROWS * 4;
    int*   widx  = (int*)w;   w += (size_t)EF_ROWS * 4;
    float* dist  = (float*)w; w += (size_t)EF_ROWS * CODEBOOK_SZ * 4;                    // 64 MB

    const int T = 256;
    // 1) converts + codebook norms
    {
        int n = BATCH * HIDDEN_DIM;
        k_f32_to_bf16<<<(n + T - 1) / T, T, 0, stream>>>(x, xb, n);
    }
    {
        int n = ENC_OUT * HIDDEN_DIM;
        k_f32_to_bf16<<<(n + T - 1) / T, T, 0, stream>>>(enc_w, ewb, n);
    }
    {
        int n = HIDDEN_DIM * ENC_OUT;
        k_f32_to_bf16<<<(n + T - 1) / T, T, 0, stream>>>(dec_w, dwb, n);
    }
    {
        int n = CODEBOOK_SZ * CODE_DIM;
        k_f32_to_bf16<<<(n + T - 1) / T, T, 0, stream>>>(codebook, cbb, n);
        k_rownorm<<<(CODEBOOK_SZ + T - 1) / T, T, 0, stream>>>(codebook, cnorm,
                                                               CODEBOOK_SZ, CODE_DIM);
    }
    // 2) encoder GEMM (TDM-staged WMMA): e = x . enc_w^T + enc_b   [512,2048]
    k_gemm_bf16_nt_tdm<<<dim3(ENC_OUT / 64, BATCH / 32), 32, 0, stream>>>(
        xb, ewb, enc_b, e_out, BATCH, ENC_OUT, HIDDEN_DIM);
    // 3) e norms + bf16(e)
    {
        int n = EF_ROWS * CODE_DIM;
        k_f32_to_bf16<<<(n + T - 1) / T, T, 0, stream>>>(e_out, ebuf, n);
        k_rownorm<<<(EF_ROWS + T - 1) / T, T, 0, stream>>>(e_out, enorm,
                                                           EF_ROWS, CODE_DIM);
    }
    // 4) distance matrix [16384,1024]
    k_dist<<<dim3(CODEBOOK_SZ / 32, EF_ROWS / 32), 32, 0, stream>>>(
        ebuf, cbb, enorm, cnorm, dist);
    // 5) per-sample Hungarian assignment (one wave32 per sample)
    k_hungarian<<<BATCH, 32, 0, stream>>>(dist, widx);
    // 6) gather q + export indices
    {
        int n = EF_ROWS * CODE_DIM;
        k_gather<<<(n + T - 1) / T, T, 0, stream>>>(codebook, widx, q_out, qb,
                                                    fi_out, EF_ROWS);
    }
    // 7) decoder GEMM (TDM-staged WMMA): r = q . dec_w^T + dec_b   [512,4096]
    k_gemm_bf16_nt_tdm<<<dim3(HIDDEN_DIM / 64, BATCH / 32), 32, 0, stream>>>(
        qb, dwb, dec_b, r_out, BATCH, HIDDEN_DIM, ENC_OUT);
}
